// GRUSeq2Seq_84696755077677
// MI455X (gfx1250) — matrix operand order repaired and checked by reading, not compiled, not run on hardware
//
#include <hip/hip_runtime.h>
#include <hip/hip_bf16.h>

typedef _Float16 h16;
typedef __attribute__((ext_vector_type(16))) _Float16 v16h;
typedef __attribute__((ext_vector_type(8)))  float    v8f;

union F16Frag { v16h v; uint4 q[2]; };

#define Tn 2048
#define Bn 64
#define Hn 128
#define G3 384      // 3*H
#define D2n 256     // 2*H
#define TBn (Tn*Bn) // 131072

// padded LDS strides (avoid 256B-periodic bank aliasing)
#define WHH_STRIDE 136   // halves per w_hh row (128 data + 8 pad)
#define H_STRIDE   136   // halves per h row
#define HG_STRIDE  392   // floats per hg row (384 data + 8 pad)

// LDS partition byte offsets for gru_rec
#define SW_OFF  0
#define SH_OFF  (384 * WHH_STRIDE * 2)                 // 104448
#define SG_OFF  (SH_OFF + Bn * H_STRIDE * 2)           // +17408 = 121856
#define SX_OFF  (SG_OFF + Bn * HG_STRIDE * 4)          // +100352 = 222208
#define SMEM_BYTES (SX_OFF + Bn * G3 * 4)              // +98304 = 320512

// ---- fast activations on the CDNA5 hardware tanh unit -------------------
__device__ __forceinline__ float tanh_hw(float x) {
#if __has_builtin(__builtin_amdgcn_tanhf)
    return __builtin_amdgcn_tanhf(x);       // v_tanh_f32 (gfx1250 TRANS op)
#else
    float e = __expf(2.0f * x);
    return (e - 1.0f) / (e + 1.0f);
#endif
}
__device__ __forceinline__ float sigm(float x) {
    return fmaf(0.5f, tanh_hw(0.5f * x), 0.5f);   // 1 TRANS + FMA
}

typedef __attribute__((address_space(3))) char lds_char;
__device__ __forceinline__ unsigned lds_addr_of(void* p) {
    return (unsigned)(unsigned long long)(lds_char*)p;
}

// issue async copy of one xg_t slice (Bn*G3 floats = 98304 B) into LDS stage.
// 256 threads x 24 B128 chunks; tracked by ASYNCcnt, overlapped with the WMMA GEMM.
__device__ __forceinline__ void prefetch_xg_async(unsigned sx_base, const float* xgt, int tid) {
    unsigned long long g = (unsigned long long)xgt;
    unsigned voff = (unsigned)tid * 16u;
#pragma unroll
    for (int k = 0; k < 24; k++) {
        unsigned o = voff + (unsigned)k * 4096u;
        asm volatile("global_load_async_to_lds_b128 %0, %1, %2"
                     :: "v"(sx_base + o), "v"(o), "s"(g) : "memory");
    }
}

__device__ __forceinline__ void wait_async0() {
#if __has_builtin(__builtin_amdgcn_s_wait_asynccnt)
    __builtin_amdgcn_s_wait_asynccnt(0);
#else
    asm volatile("s_wait_asynccnt 0x0" ::: "memory");
#endif
}

// ---------------------------------------------------------------- f32 -> f16
__global__ void cvt_f32_f16(h16* __restrict__ dst, const float* __restrict__ src, int n) {
    int i = blockIdx.x * 256 + threadIdx.x;
    if (i < n) dst[i] = (h16)src[i];
}

// ------------------------------------------------- enc0 projection (K = 7)
__global__ void proj_in7(const float* __restrict__ x, const float* __restrict__ wih,
                         const float* __restrict__ bih, float* __restrict__ xg) {
    long i = (long)blockIdx.x * 256 + threadIdx.x;
    const long total = 2L * Tn * Bn * G3;
    if (i >= total) return;
    int g = (int)(i % G3); long r = i / G3;
    int b = (int)(r % Bn); r /= Bn;
    int t = (int)(r % Tn); int d = (int)(r / Tn);
    const float* w  = wih + ((long)d * G3 + g) * 7;
    const float* xp = x + ((long)b * Tn + t) * 7;
    float acc = bih[d * G3 + g];
#pragma unroll
    for (int k = 0; k < 7; k++) acc += xp[k] * w[k];
    xg[i] = acc;
}

// ------------------------------------------- WMMA projection (in = 256)
__global__ void __launch_bounds__(256)
proj_wmma(const h16* __restrict__ A, const h16* __restrict__ wih,
          const float* __restrict__ bih, float* __restrict__ xg) {
    const int MT = TBn / 16;                       // 8192 row tiles
    int wave = (blockIdx.x * blockDim.x + threadIdx.x) >> 5;
    int lane = threadIdx.x & 31;
    int totalTiles = 2 * MT * 24;
    if (wave >= totalTiles) return;                // wave-uniform
    int nt = wave % 24; int r = wave / 24;
    int mt = r % MT;    int d = r / MT;
    int lo = lane & 15, hi = lane >> 4;
    int row0 = mt * 16;
    int col  = nt * 16 + lo;

    v8f acc;
    float bz = bih[d * G3 + col];
#pragma unroll
    for (int v = 0; v < 8; v++) acc[v] = bz;

    const h16* Arow = A + (long)(row0 + lo) * 256;
    const h16* Brow = wih + ((long)d * G3 + col) * 256;
#pragma unroll
    for (int kb = 0; kb < 256; kb += 32) {
        F16Frag a, b;
        a.q[0] = *(const uint4*)(Arow + kb + 8 * hi);
        a.q[1] = *(const uint4*)(Arow + kb + 16 + 8 * hi);
        b.q[0] = *(const uint4*)(Brow + kb + 8 * hi);
        b.q[1] = *(const uint4*)(Brow + kb + 16 + 8 * hi);
        acc = __builtin_amdgcn_wmma_f32_16x16x32_f16(false, a.v, false, b.v,
                                                     (short)0, acc, false, false);
    }
    float* out = xg + (long)d * TBn * G3;
#pragma unroll
    for (int v = 0; v < 8; v++) {
        int rrow = row0 + v + 8 * hi;
        out[(long)rrow * G3 + col] = acc[v];
    }
}

// ------------------------------------------------------- GRU recurrence
// One block per direction. 256 threads = 8 waves. W_hh + h + hg + xg-stage in LDS.
__global__ void __launch_bounds__(256)
gru_rec(const h16* __restrict__ whh, const float* __restrict__ bhh,
        const float* __restrict__ xg, const float* __restrict__ h0,
        h16* __restrict__ y, float* __restrict__ hfin) {
    extern __shared__ char smem[];
    h16*   sW = (h16*)(smem + SW_OFF);     // [384][WHH_STRIDE]
    h16*   sH = (h16*)(smem + SH_OFF);     // [64][H_STRIDE]
    float* sG = (float*)(smem + SG_OFF);   // [64][HG_STRIDE]
    float* sX = (float*)(smem + SX_OFF);   // [64][384] staged xg_t

    const int dir = blockIdx.x;
    const int tid = threadIdx.x;
    const unsigned sx_base = lds_addr_of(sX);

    // stage w_hh into LDS (16B chunks)
    for (int i = tid; i < 384 * 16; i += 256) {
        int rrow = i >> 4, c = (i & 15) * 8;
        *(uint4*)(sW + rrow * WHH_STRIDE + c) =
            *(const uint4*)(whh + ((long)dir * 384 + rrow) * 128 + c);
    }
    // init h
    for (int i = tid; i < Bn * Hn; i += 256) {
        int b = i >> 7, j = i & 127;
        float hv = h0 ? h0[((long)dir * Bn + b) * Hn + j] : 0.0f;
        sH[b * H_STRIDE + j] = (h16)hv;
    }

    const int lane = tid & 31, wave = tid >> 5;
    const int lo = lane & 15, hi = lane >> 4;

    // hoist per-tile b_hh bias into registers (loop-invariant over all 2048 steps)
    float bias[12];
#pragma unroll
    for (int i = 0; i < 12; i++) {
        int tile = wave + 8 * i;
        int nt = tile >> 2;
        bias[i] = bhh[dir * G3 + nt * 16 + lo];
    }

    // prefetch xg for step 0
    {
        int t0 = dir ? (Tn - 1) : 0;
        prefetch_xg_async(sx_base, xg + ((long)dir * Tn + t0) * Bn * G3, tid);
    }
    __syncthreads();

    for (int s = 0; s < Tn; s++) {
        int t = dir ? (Tn - 1 - s) : s;
        // hg = h @ w_hh^T + b_hh  (96 tiles: 4 M-tiles x 24 N-tiles, 12 per wave)
        // async xg copy for this step is in flight underneath this GEMM.
#pragma unroll
        for (int i = 0; i < 12; i++) {
            int tile = wave + 8 * i;
            int mt = tile & 3, nt = tile >> 2;
            int col = nt * 16 + lo;
            v8f acc;
#pragma unroll
            for (int v = 0; v < 8; v++) acc[v] = bias[i];
            const h16* hrow = sH + (mt * 16 + lo) * H_STRIDE;
            const h16* wrow = sW + col * WHH_STRIDE;
#pragma unroll
            for (int kb = 0; kb < 128; kb += 32) {
                F16Frag a, b;
                a.q[0] = *(const uint4*)(hrow + kb + 8 * hi);
                a.q[1] = *(const uint4*)(hrow + kb + 16 + 8 * hi);
                b.q[0] = *(const uint4*)(wrow + kb + 8 * hi);
                b.q[1] = *(const uint4*)(wrow + kb + 16 + 8 * hi);
                acc = __builtin_amdgcn_wmma_f32_16x16x32_f16(false, a.v, false, b.v,
                                                             (short)0, acc, false, false);
            }
#pragma unroll
            for (int v = 0; v < 8; v++) {
                int brow = mt * 16 + v + 8 * hi;
                sG[brow * HG_STRIDE + col] = acc[v];
            }
        }
        wait_async0();       // this wave's xg chunks have landed in LDS
        __syncthreads();     // + everyone else's; sG complete too

        // fused gates: 3 hardware-tanh TRANS ops per element, xg from LDS stage
        for (int i = tid; i < Bn * Hn; i += 256) {
            int b = i >> 7, j = i & 127;
            const float* xb = sX + b * G3;
            float hr = sG[b * HG_STRIDE + j];
            float hz = sG[b * HG_STRIDE + 128 + j];
            float hn = sG[b * HG_STRIDE + 256 + j];
            float rg = sigm(xb[j] + hr);
            float zg = sigm(xb[128 + j] + hz);
            float ng = tanh_hw(xb[256 + j] + rg * hn);
            float hp = (float)sH[b * H_STRIDE + j];
            float hnew = (1.0f - zg) * ng + zg * hp;
            sH[b * H_STRIDE + j] = (h16)hnew;
            y[((long)t * Bn + b) * D2n + dir * Hn + j] = (h16)hnew;
        }
        __syncthreads();     // sX/sG/sH consumption complete

        if (s + 1 < Tn) {    // launch next step's xg copy under next GEMM
            int tn = dir ? (Tn - 2 - s) : (s + 1);
            prefetch_xg_async(sx_base, xg + ((long)dir * Tn + tn) * Bn * G3, tid);
        }
    }

    if (hfin) {
        for (int i = tid; i < Bn * Hn; i += 256) {
            int b = i >> 7, j = i & 127;
            hfin[((long)dir * Bn + b) * Hn + j] = (float)sH[b * H_STRIDE + j];
        }
    }
}

// ------------------------------------------------------------ FC head
__global__ void fc_out(const h16* __restrict__ y, const float* __restrict__ fcw,
                       const float* __restrict__ fcb, float* __restrict__ out) {
    long i = (long)blockIdx.x * 256 + threadIdx.x;   // over T*B
    if (i >= (long)TBn) return;
    int b = (int)(i % Bn); int t = (int)(i / Bn);
    const h16* yp = y + i * D2n;
    float acc = fcb[0];
#pragma unroll 8
    for (int j = 0; j < D2n; j++) acc += (float)yp[j] * fcw[j];
    out[(long)b * Tn + t] = acc;
}

extern "C" void kernel_launch(void* const* d_in, const int* in_sizes, int n_in,
                              void* d_out, int out_size, void* d_ws, size_t ws_size,
                              hipStream_t stream) {
    const float* x       = (const float*)d_in[0];
    const float* wih_f[4] = {(const float*)d_in[1], (const float*)d_in[5],
                             (const float*)d_in[9], (const float*)d_in[13]};
    const float* whh_f[4] = {(const float*)d_in[2], (const float*)d_in[6],
                             (const float*)d_in[10], (const float*)d_in[14]};
    const float* bih_f[4] = {(const float*)d_in[3], (const float*)d_in[7],
                             (const float*)d_in[11], (const float*)d_in[15]};
    const float* bhh_f[4] = {(const float*)d_in[4], (const float*)d_in[8],
                             (const float*)d_in[12], (const float*)d_in[16]};
    const float* fcw = (const float*)d_in[17];
    const float* fcb = (const float*)d_in[18];
    float* out = (float*)d_out;

    // workspace partition (256B aligned chunks)
    char* p = (char*)d_ws;
    auto take = [&](size_t bytes) { char* r = p; p += (bytes + 255) & ~(size_t)255; return r; };
    const size_t WIH_H = 2UL * G3 * 256;   // halves per 256-input layer
    const size_t WHH_H = 2UL * G3 * Hn;    // halves per layer
    h16*   wih_h = (h16*)take(3 * WIH_H * sizeof(h16));   // enc1, dec0, dec1
    h16*   whh_h = (h16*)take(4 * WHH_H * sizeof(h16));   // enc0..dec1
    h16*   yA    = (h16*)take((size_t)TBn * D2n * sizeof(h16));
    h16*   yB    = (h16*)take((size_t)TBn * D2n * sizeof(h16));
    float* xg    = (float*)take(2UL * TBn * G3 * sizeof(float));
    float* hfin0 = (float*)take(2UL * Bn * Hn * sizeof(float));
    float* hfin1 = (float*)take(2UL * Bn * Hn * sizeof(float));

    // weight conversion
    for (int l = 1; l < 4; l++)
        cvt_f32_f16<<<(int)((WIH_H + 255) / 256), 256, 0, stream>>>(
            wih_h + (l - 1) * WIH_H, wih_f[l], (int)WIH_H);
    for (int l = 0; l < 4; l++)
        cvt_f32_f16<<<(int)((WHH_H + 255) / 256), 256, 0, stream>>>(
            whh_h + l * WHH_H, whh_f[l], (int)WHH_H);

    const long projSmallN = 2L * Tn * Bn * G3;
    const int  projGrid   = (2 * (TBn / 16) * 24) / 8;   // 8 waves per block

    // enc0
    proj_in7<<<(int)((projSmallN + 255) / 256), 256, 0, stream>>>(x, wih_f[0], bih_f[0], xg);
    gru_rec<<<2, 256, SMEM_BYTES, stream>>>(whh_h + 0 * WHH_H, bhh_f[0], xg, nullptr, yA, hfin0);
    // enc1
    proj_wmma<<<projGrid, 256, 0, stream>>>(yA, wih_h + 0 * WIH_H, bih_f[1], xg);
    gru_rec<<<2, 256, SMEM_BYTES, stream>>>(whh_h + 1 * WHH_H, bhh_f[1], xg, nullptr, yB, hfin1);
    // dec0 (seeded with enc0 final hidden)
    proj_wmma<<<projGrid, 256, 0, stream>>>(yB, wih_h + 1 * WIH_H, bih_f[2], xg);
    gru_rec<<<2, 256, SMEM_BYTES, stream>>>(whh_h + 2 * WHH_H, bhh_f[2], xg, hfin0, yA, nullptr);
    // dec1 (seeded with enc1 final hidden)
    proj_wmma<<<projGrid, 256, 0, stream>>>(yA, wih_h + 2 * WIH_H, bih_f[3], xg);
    gru_rec<<<2, 256, SMEM_BYTES, stream>>>(whh_h + 3 * WHH_H, bhh_f[3], xg, hfin1, yB, nullptr);
    // head
    fc_out<<<(TBn + 255) / 256, 256, 0, stream>>>(yB, fcw, fcb, out);
}
